// Architecture1_60103772340713
// MI455X (gfx1250) — matrix operand order repaired and checked by reading, not compiled, not run on hardware
//
#include <hip/hip_runtime.h>
#include <stdint.h>

// Model dims
#define Bb 32
#define Tt 64
#define Ee 512
#define Hh 1024
#define G4H 4096
#define Vv 32000
#define BT 2048
#define BN_EPS 1e-5f

typedef __attribute__((ext_vector_type(16))) __bf16 v16bf;
typedef __attribute__((ext_vector_type(8)))  float  v8f;

struct Frag16 { union { v16bf v; uint32_t u[8]; uint4 q[2]; }; };

static __device__ __forceinline__ uint32_t pack2(float a, float b) {
    union { __bf16 h[2]; uint32_t u; } p;
    p.h[0] = (__bf16)a; p.h[1] = (__bf16)b;
    return p.u;
}

// A-fragment 16x32 bf16 (row-major A[M][K], K-contiguous). contiguous k order used here (B matches):
// lane<16: M=row0+lane, K = 0..15; lane>=16: M=row0+lane-16, K = 16..31
static __device__ __forceinline__ Frag16 load_a_frag(const __bf16* A, int lda,
                                                     int row0, int k0, int lane) {
    Frag16 f;
    const int half = lane >> 4;
    const int m = row0 + (lane & 15);
    const __bf16* p = A + (size_t)m * lda + k0 + half * 16;
    f.q[0] = *reinterpret_cast<const uint4*>(p);        // K: half*16 .. half*16+7
    f.q[1] = *reinterpret_cast<const uint4*>(p + 8);    // K: half*16+8 ..
    return f;
}

// B-fragment 32x16 bf16 from NT weight W[N][K] (K-contiguous). contiguous k order used here (A matches):
// lanes 0-15 hold K=0..15 of column N=col0+lane; lanes 16-31 hold K=16..31
static __device__ __forceinline__ Frag16 load_b_frag_bf16(const __bf16* W, int ldw,
                                                          int col0, int k0, int lane) {
    Frag16 f;
    const int half = lane >> 4;
    const int n = col0 + (lane & 15);
    const __bf16* p = W + (size_t)n * ldw + k0 + half * 16;
    f.q[0] = *reinterpret_cast<const uint4*>(p);        // K: +0..7
    f.q[1] = *reinterpret_cast<const uint4*>(p + 8);    // K: +8..15
    return f;
}

// Same B-fragment but converting fp32 weights to bf16 on the fly (fc_W path)
static __device__ __forceinline__ Frag16 load_b_frag_f32(const float* W, int ldw,
                                                         int col0, int k0, int lane) {
    Frag16 f;
    const int half = lane >> 4;
    const int n = col0 + (lane & 15);
    const float* p = W + (size_t)n * ldw + k0 + half * 16;
    float4 x0 = reinterpret_cast<const float4*>(p)[0];
    float4 x1 = reinterpret_cast<const float4*>(p)[1];
    float4 x2 = reinterpret_cast<const float4*>(p)[2];
    float4 x3 = reinterpret_cast<const float4*>(p)[3];
    f.u[0] = pack2(x0.x, x0.y); f.u[1] = pack2(x0.z, x0.w);
    f.u[2] = pack2(x1.x, x1.y); f.u[3] = pack2(x1.z, x1.w);
    f.u[4] = pack2(x2.x, x2.y); f.u[5] = pack2(x2.z, x2.w);
    f.u[6] = pack2(x3.x, x3.y); f.u[7] = pack2(x3.z, x3.w);
    return f;
}

#define WMMA_BF16(A_, B_, C_) __builtin_amdgcn_wmma_f32_16x16x32_bf16( \
    false, (A_), false, (B_), (short)0, (C_), false, false)

// ---------------- Generic NT GEMM: C[M][N] = A[M][K] * W[N][K]^T + bias1 (+bias2) ----------
// block = 256 threads (8 waves), block tile 64(M) x 128(N), each wave 16x64.
// K-loop is software-pipelined: next step's 10 b128 loads issue before current 4 WMMAs.
__global__ __launch_bounds__(256)
void gemm_nt_bias(const __bf16* __restrict__ A, const __bf16* __restrict__ W,
                  const float* __restrict__ bias1, const float* __restrict__ bias2,
                  float* __restrict__ C, int N, int K) {
    const int lane = threadIdx.x & 31;
    const int w    = threadIdx.x >> 5;
    const int row0 = blockIdx.x * 64 + (w & 3) * 16;
    const int colb = blockIdx.y * 128 + (w >> 2) * 64;

    v8f acc[4] = {};
    Frag16 a = load_a_frag(A, K, row0, 0, lane);
    Frag16 b[4];
#pragma unroll
    for (int tI = 0; tI < 4; ++tI) b[tI] = load_b_frag_bf16(W, K, colb + tI * 16, 0, lane);

    for (int k0 = 32; k0 < K; k0 += 32) {
        Frag16 an = load_a_frag(A, K, row0, k0, lane);
        Frag16 bn[4];
#pragma unroll
        for (int tI = 0; tI < 4; ++tI)
            bn[tI] = load_b_frag_bf16(W, K, colb + tI * 16, k0, lane);
#pragma unroll
        for (int tI = 0; tI < 4; ++tI)
            acc[tI] = WMMA_BF16(a.v, b[tI].v, acc[tI]);
        a = an;
#pragma unroll
        for (int tI = 0; tI < 4; ++tI) b[tI] = bn[tI];
    }
#pragma unroll
    for (int tI = 0; tI < 4; ++tI)
        acc[tI] = WMMA_BF16(a.v, b[tI].v, acc[tI]);

    // C/D layout: lane<16 -> M=row0+r, N=col+lane ; lane>=16 -> M=row0+8+r
    const int mbase = row0 + (lane >> 4) * 8;
#pragma unroll
    for (int tI = 0; tI < 4; ++tI) {
        const int n = colb + tI * 16 + (lane & 15);
        float bs = bias1[n] + (bias2 ? bias2[n] : 0.0f);
#pragma unroll
        for (int r = 0; r < 8; ++r)
            C[(size_t)(mbase + r) * N + n] = acc[tI][r] + bs;
    }
}

// ---------------- Final FC: out[2048][32000] = hs1 * fc_W^T + fc_b (fp32 W, cvt on the fly)
__global__ __launch_bounds__(256)
void fc_gemm(const __bf16* __restrict__ A, const float* __restrict__ W,
             const float* __restrict__ bias, float* __restrict__ C) {
    const int lane = threadIdx.x & 31;
    const int w    = threadIdx.x >> 5;
    const int row0 = blockIdx.x * 64 + (w & 3) * 16;
    const int colb = blockIdx.y * 128 + (w >> 2) * 64;

    v8f acc[4] = {};
    Frag16 a = load_a_frag(A, Hh, row0, 0, lane);
    Frag16 b[4];
#pragma unroll
    for (int tI = 0; tI < 4; ++tI) b[tI] = load_b_frag_f32(W, Hh, colb + tI * 16, 0, lane);

    for (int k0 = 32; k0 < Hh; k0 += 32) {
        Frag16 an = load_a_frag(A, Hh, row0, k0, lane);
        Frag16 bn[4];
#pragma unroll
        for (int tI = 0; tI < 4; ++tI)
            bn[tI] = load_b_frag_f32(W, Hh, colb + tI * 16, k0, lane);
#pragma unroll
        for (int tI = 0; tI < 4; ++tI)
            acc[tI] = WMMA_BF16(a.v, b[tI].v, acc[tI]);
        a = an;
#pragma unroll
        for (int tI = 0; tI < 4; ++tI) b[tI] = bn[tI];
    }
#pragma unroll
    for (int tI = 0; tI < 4; ++tI)
        acc[tI] = WMMA_BF16(a.v, b[tI].v, acc[tI]);

    const int mbase = row0 + (lane >> 4) * 8;
#pragma unroll
    for (int tI = 0; tI < 4; ++tI) {
        const int n = colb + tI * 16 + (lane & 15);
        float bs = bias[n];
#pragma unroll
        for (int r = 0; r < 8; ++r)
            C[(size_t)(mbase + r) * Vv + n] = acc[tI][r] + bs;
    }
}

// ---------------- Fused LSTM recurrent step -------------------------------------------------
// g = pre[:,t,:] + h_in @ Whh^T ; gates -> update c (f32), emit h (bf16) to h_out and hs[:,t,:]
// grid = 16 blocks x 256 threads = 128 waves; wave -> (m_tile in {0,1}, j_tile in 0..63),
// wave computes the 4 gate tiles {i,f,g,o} for its 16-column j range (A-frag reused 4x).
__global__ __launch_bounds__(256)
void lstm_step(const float* __restrict__ pre, const __bf16* __restrict__ h_in,
               const __bf16* __restrict__ Whh, float* __restrict__ c_st,
               __bf16* __restrict__ h_out, __bf16* __restrict__ hs, int t) {
    const int lane = threadIdx.x & 31;
    const int gw   = blockIdx.x * 8 + (threadIdx.x >> 5);
    const int row0 = (gw & 1) * 16;        // batch rows 0..31
    const int j0   = (gw >> 1) * 16;       // hidden unit tile

    v8f acc[4] = {};
    Frag16 a = load_a_frag(h_in, Hh, row0, 0, lane);
    Frag16 b[4];
#pragma unroll
    for (int g = 0; g < 4; ++g) b[g] = load_b_frag_bf16(Whh, Hh, g * Hh + j0, 0, lane);

    for (int k0 = 32; k0 < Hh; k0 += 32) {
        Frag16 an = load_a_frag(h_in, Hh, row0, k0, lane);
        Frag16 bn[4];
#pragma unroll
        for (int g = 0; g < 4; ++g)
            bn[g] = load_b_frag_bf16(Whh, Hh, g * Hh + j0, k0, lane);
#pragma unroll
        for (int g = 0; g < 4; ++g)
            acc[g] = WMMA_BF16(a.v, b[g].v, acc[g]);
        a = an;
#pragma unroll
        for (int g = 0; g < 4; ++g) b[g] = bn[g];
    }
#pragma unroll
    for (int g = 0; g < 4; ++g)
        acc[g] = WMMA_BF16(a.v, b[g].v, acc[g]);

    const int mbase = row0 + (lane >> 4) * 8;
    const int j = j0 + (lane & 15);
#pragma unroll
    for (int r = 0; r < 8; ++r) {
        const int b_ = mbase + r;                      // batch index
        const size_t prow = ((size_t)b_ * Tt + t) * (size_t)G4H;
        float gi = acc[0][r] + pre[prow + 0 * Hh + j];
        float gf = acc[1][r] + pre[prow + 1 * Hh + j];
        float gg = acc[2][r] + pre[prow + 2 * Hh + j];
        float go = acc[3][r] + pre[prow + 3 * Hh + j];
        float si = 1.0f / (1.0f + expf(-gi));
        float sf = 1.0f / (1.0f + expf(-gf));
        float so = 1.0f / (1.0f + expf(-go));
        const size_t cidx = (size_t)b_ * Hh + j;
        float cn = sf * c_st[cidx] + si * tanhf(gg);
        float hn = so * tanhf(cn);
        c_st[cidx] = cn;
        __bf16 hb = (__bf16)hn;
        h_out[cidx] = hb;
        hs[((size_t)b_ * Tt + t) * Hh + j] = hb;
    }
}

// ---------------- Small stem / utility kernels ---------------------------------------------
__global__ void stem_linear(const float* __restrict__ img, const float* __restrict__ W,
                            const float* __restrict__ b, float* __restrict__ x) {
    const int tid = blockIdx.x * blockDim.x + threadIdx.x;   // 16384 = 32*512
    const int bi = tid >> 9, e = tid & 511;
    const float4* af = reinterpret_cast<const float4*>(img + (size_t)bi * 2048);
    const float4* wf = reinterpret_cast<const float4*>(W + (size_t)e * 2048);
    float acc = 0.0f;
    for (int k = 0; k < 512; ++k) {
        float4 a = af[k], wv = wf[k];
        acc += a.x * wv.x + a.y * wv.y + a.z * wv.z + a.w * wv.w;
    }
    x[tid] = acc + b[e];
}

__global__ void bn_slot0(const float* __restrict__ x, const float* __restrict__ gamma,
                         const float* __restrict__ beta, __bf16* __restrict__ xs) {
    const int e = blockIdx.x * blockDim.x + threadIdx.x;     // 0..511
    float s = 0.0f, s2 = 0.0f;
    for (int b = 0; b < Bb; ++b) { float v = x[b * Ee + e]; s += v; s2 += v * v; }
    const float mu = s * (1.0f / Bb);
    const float var = s2 * (1.0f / Bb) - mu * mu;
    const float inv = rsqrtf(var + BN_EPS);
    const float ga = gamma[e], be = beta[e];
    for (int b = 0; b < Bb; ++b) {
        float v = ga * (x[b * Ee + e] - mu) * inv + be;
        xs[((size_t)b * Tt) * Ee + e] = (__bf16)v;           // t = 0 slot
    }
}

__global__ void embed_gather(const int* __restrict__ cap, const float* __restrict__ emb,
                             __bf16* __restrict__ xs) {
    const int row = blockIdx.x;            // 0..2015  (b in 0..31, tt in 0..62)
    const int b = row / 63, tt = row % 63;
    const int t = tt + 1;
    const int idx = cap[b * Tt + tt];      // captions[:, :-1]
    const int e0 = threadIdx.x * 4;        // 128 threads * 4 = 512
    float4 v = *reinterpret_cast<const float4*>(emb + (size_t)idx * Ee + e0);
    uint2 o; o.x = pack2(v.x, v.y); o.y = pack2(v.z, v.w);
    *reinterpret_cast<uint2*>(xs + ((size_t)b * Tt + t) * Ee + e0) = o;
}

__global__ void cvt_f32_bf16(const float* __restrict__ src, __bf16* __restrict__ dst, int n8) {
    const int i = blockIdx.x * blockDim.x + threadIdx.x;
    if (i >= n8) return;
    const float4* s = reinterpret_cast<const float4*>(src) + (size_t)i * 2;
    float4 a = s[0], b = s[1];
    uint4 o;
    o.x = pack2(a.x, a.y); o.y = pack2(a.z, a.w);
    o.z = pack2(b.x, b.y); o.w = pack2(b.z, b.w);
    reinterpret_cast<uint4*>(dst)[i] = o;
}

__global__ void zero_state(float* __restrict__ c, __bf16* __restrict__ h) {
    const int i = blockIdx.x * blockDim.x + threadIdx.x;     // 32768 = 32*1024
    c[i] = 0.0f;
    h[i] = (__bf16)0.0f;
}

// ---------------- Host launcher -------------------------------------------------------------
extern "C" void kernel_launch(void* const* d_in, const int* in_sizes, int n_in,
                              void* d_out, int out_size, void* d_ws, size_t ws_size,
                              hipStream_t stream) {
    const float* image_feat = (const float*)d_in[0];
    const int*   captions   = (const int*)  d_in[1];
    const float* lin_W      = (const float*)d_in[2];
    const float* lin_b      = (const float*)d_in[3];
    const float* bn_gamma   = (const float*)d_in[4];
    const float* bn_beta    = (const float*)d_in[5];
    const float* emb        = (const float*)d_in[6];
    const float* Wih0       = (const float*)d_in[7];
    const float* Whh0       = (const float*)d_in[8];
    const float* bih0       = (const float*)d_in[9];
    const float* bhh0       = (const float*)d_in[10];
    const float* Wih1       = (const float*)d_in[11];
    const float* Whh1       = (const float*)d_in[12];
    const float* bih1       = (const float*)d_in[13];
    const float* bhh1       = (const float*)d_in[14];
    const float* fc_W       = (const float*)d_in[15];
    const float* fc_b       = (const float*)d_in[16];
    float* out = (float*)d_out;

    // Workspace carve (256B aligned), total ~74 MB
    size_t off = 0;
    auto carve = [&](size_t bytes) {
        void* p = (char*)d_ws + off;
        off += (bytes + 255) & ~(size_t)255;
        return p;
    };
    float*  x_f32   = (float*) carve((size_t)Bb * Ee * 4);         // 64 KB
    __bf16* xs      = (__bf16*)carve((size_t)BT * Ee * 2);         // 2 MB
    float*  pre     = (float*) carve((size_t)BT * G4H * 4);        // 32 MB (reused L0/L1)
    __bf16* hs0     = (__bf16*)carve((size_t)BT * Hh * 2);         // 4 MB
    __bf16* hs1     = (__bf16*)carve((size_t)BT * Hh * 2);         // 4 MB
    float*  c_st    = (float*) carve((size_t)Bb * Hh * 4);         // 128 KB
    __bf16* hbufA   = (__bf16*)carve((size_t)Bb * Hh * 2);         // 64 KB
    __bf16* hbufB   = (__bf16*)carve((size_t)Bb * Hh * 2);         // 64 KB
    __bf16* Wih0_bf = (__bf16*)carve((size_t)G4H * Ee * 2);        // 4 MB
    __bf16* Whh0_bf = (__bf16*)carve((size_t)G4H * Hh * 2);        // 8 MB
    __bf16* Wih1_bf = (__bf16*)carve((size_t)G4H * Hh * 2);        // 8 MB
    __bf16* Whh1_bf = (__bf16*)carve((size_t)G4H * Hh * 2);        // 8 MB

    // 1) weight conversions to bf16
    {
        int n8 = (G4H * Ee) / 8;
        cvt_f32_bf16<<<(n8 + 255) / 256, 256, 0, stream>>>(Wih0, Wih0_bf, n8);
        n8 = (G4H * Hh) / 8;
        cvt_f32_bf16<<<(n8 + 255) / 256, 256, 0, stream>>>(Whh0, Whh0_bf, n8);
        cvt_f32_bf16<<<(n8 + 255) / 256, 256, 0, stream>>>(Wih1, Wih1_bf, n8);
        cvt_f32_bf16<<<(n8 + 255) / 256, 256, 0, stream>>>(Whh1, Whh1_bf, n8);
    }

    // 2) stem: linear -> batchnorm -> xs[:,0,:]; embeddings -> xs[:,1:,:]
    stem_linear<<<(Bb * Ee) / 256, 256, 0, stream>>>(image_feat, lin_W, lin_b, x_f32);
    bn_slot0<<<Ee / 256, 256, 0, stream>>>(x_f32, bn_gamma, bn_beta, xs);
    embed_gather<<<Bb * (Tt - 1), 128, 0, stream>>>(captions, emb, xs);

    // 3) layer 0: input projection + recurrence
    {
        dim3 grid(BT / 64, G4H / 128);
        gemm_nt_bias<<<grid, 256, 0, stream>>>(xs, Wih0_bf, bih0, bhh0, pre, G4H, Ee);
    }
    zero_state<<<(Bb * Hh) / 256, 256, 0, stream>>>(c_st, hbufA);
    for (int t = 0; t < Tt; ++t) {
        const __bf16* hin  = (t & 1) ? hbufB : hbufA;
        __bf16*       hout = (t & 1) ? hbufA : hbufB;
        lstm_step<<<16, 256, 0, stream>>>(pre, hin, Whh0_bf, c_st, hout, hs0, t);
    }

    // 4) layer 1: input projection + recurrence (reuse `pre`)
    {
        dim3 grid(BT / 64, G4H / 128);
        gemm_nt_bias<<<grid, 256, 0, stream>>>(hs0, Wih1_bf, bih1, bhh1, pre, G4H, Hh);
    }
    zero_state<<<(Bb * Hh) / 256, 256, 0, stream>>>(c_st, hbufA);
    for (int t = 0; t < Tt; ++t) {
        const __bf16* hin  = (t & 1) ? hbufB : hbufA;
        __bf16*       hout = (t & 1) ? hbufA : hbufB;
        lstm_step<<<16, 256, 0, stream>>>(pre, hin, Whh1_bf, c_st, hout, hs1, t);
    }

    // 5) final FC to vocabulary (dominant GEMM; fc_W stays L2-resident)
    {
        dim3 grid(BT / 64, Vv / 128);
        fc_gemm<<<grid, 256, 0, stream>>>(hs1, fc_W, fc_b, out);
    }
    (void)in_sizes; (void)n_in; (void)out_size; (void)ws_size;
}
